// Qwen3VLVisionAttention_28578712387606
// MI455X (gfx1250) — matrix operand order repaired and checked by reading, not compiled, not run on hardware
//
#include <hip/hip_runtime.h>

typedef __attribute__((ext_vector_type(16))) __bf16 v16bf;
typedef __attribute__((ext_vector_type(8)))  __bf16 v8bf;
typedef __attribute__((ext_vector_type(4)))  __bf16 v4bf;
typedef __attribute__((ext_vector_type(8)))  float  v8f;
typedef __attribute__((ext_vector_type(4)))  unsigned int u32x4;
typedef __attribute__((ext_vector_type(8)))  int        i32x8;
typedef __attribute__((ext_vector_type(4)))  int        i32x4;

#define S_TOT   4096
#define E_DIM   1152
#define N_QKV   3456
#define HQ      16
#define DH      72
#define DPAD    96
#define DV      80
#define LCHUNK  1024
#define ATT_SCALE 0.117851130197757924f   /* 72^-0.5 */

#if defined(__gfx1250__) && __has_builtin(__builtin_amdgcn_tensor_load_to_lds)
#define HAVE_TDM 1
#endif

__device__ __forceinline__ v8f wmma_bf16(v16bf a, v16bf b, v8f c) {
  return __builtin_amdgcn_wmma_f32_16x16x32_bf16(false, a, false, b, (short)0, c, false, false);
}

// ---- Tensor Data Mover: 2D tile load (bf16 elements), optional LDS padding.
// pad_int_code: pads inserted every {2,4,8,16,32,64,128,256}[code] DWORDs stored
// pad_amt_code: pad of (code+1) DWORDs
__device__ __forceinline__ void tdm_load_2d(unsigned int lds_off, const void* gaddr,
                                            unsigned int tile0, unsigned int tile1,
                                            unsigned int stride0,
                                            unsigned int pad_int_code,
                                            unsigned int pad_amt_code) {
#ifdef HAVE_TDM
  unsigned long long ga = (unsigned long long)gaddr;
  u32x4 g0;
  g0[0] = 1u;                                                  // count=1 (valid), user mode
  g0[1] = lds_off;                                             // LDS byte address
  g0[2] = (unsigned int)(ga & 0xFFFFFFFFu);                    // global addr lo
  g0[3] = (unsigned int)((ga >> 32) & 0x1FFFFFFu) | (2u << 30);// global addr hi, type=2
  i32x8 g1;
  g1[0] = (int)((1u << 16) |                  // data_size = 2 bytes
                (1u << 20) |                  // pad_enable
                (pad_int_code << 22) | (pad_amt_code << 25));
  g1[1] = (int)((tile0 & 0xFFFFu) << 16);     // tensor_dim0 = tile0 (lo16)
  g1[2] = (int)(((tile0 >> 16) & 0xFFFFu) | ((tile1 & 0xFFFFu) << 16)); // dim0 hi | dim1 lo
  g1[3] = (int)(((tile1 >> 16) & 0xFFFFu) | (tile0 << 16));   // dim1 hi | tile_dim0
  g1[4] = (int)(tile1 & 0xFFFFu);                             // tile_dim1 (tile_dim2=0)
  g1[5] = (int)stride0;                                        // tensor_dim0_stride lo32
  g1[6] = 0;                                                   // stride hi, dim1_stride lo
  g1[7] = 0;
  i32x4 z4 = {0, 0, 0, 0};
  i32x8 z8 = {0, 0, 0, 0, 0, 0, 0, 0};
  __builtin_amdgcn_tensor_load_to_lds(g0, g1, z4, z4, z8, 0);
#else
  (void)lds_off; (void)gaddr; (void)tile0; (void)tile1;
  (void)stride0; (void)pad_int_code; (void)pad_amt_code;
#endif
}

__device__ __forceinline__ void tdm_wait() {
#ifdef HAVE_TDM
  __builtin_amdgcn_s_wait_tensorcnt(0);
#endif
}

// A fragment (16x32 bf16): lane<16 row=lane, K {0..7,16..23}; lane>=16 row=lane-16, K {8..15,24..31}
__device__ __forceinline__ v16bf frag_a(const __bf16* base, int ld, int lane) {
  int r = lane & 15, hf = (lane >> 4) * 8;
  v8bf lo = *(const v8bf*)(base + r * ld + hf);
  v8bf hi = *(const v8bf*)(base + r * ld + hf + 16);
  v16bf a;
#pragma unroll
  for (int i = 0; i < 8; ++i) { a[i] = lo[i]; a[i + 8] = hi[i]; }
  return a;
}

// B fragment (32x16 bf16): lane holds column n=lane%16; lane<16 K {0..7,16..23}, lane>=16 K {8..15,24..31}
__device__ __forceinline__ v16bf frag_b(const __bf16* base, int ld, int lane) {
  int r = lane & 15, hf = (lane >> 4) * 8;
  v8bf lo = *(const v8bf*)(base + r * ld + hf);
  v8bf hi = *(const v8bf*)(base + r * ld + hf + 16);
  v16bf b;
#pragma unroll
  for (int i = 0; i < 8; ++i) { b[i] = lo[i]; b[i + 8] = hi[i]; }
  return b;
}

// ---------------- GEMM: C[M,N] = A[M,K] * W[N,K]^T + bias --------------------
#define LDT 40
template <bool AF32, bool OUTF32>
__global__ __launch_bounds__(256) void gemm_wmma(
    const float* __restrict__ Af, const unsigned short* __restrict__ Abu,
    const float* __restrict__ Wf, const float* __restrict__ bias,
    float* __restrict__ Cf, unsigned short* __restrict__ Cbu,
    int M, int N, int K) {
  __shared__ __bf16 As[128 * LDT];
  __shared__ __bf16 Bs[128 * LDT];
  const __bf16* Ab = (const __bf16*)Abu;
  __bf16* Cb = (__bf16*)Cbu;

  const int mb = blockIdx.x * 128;
  const int nb = blockIdx.y * 128;
  const int tid = threadIdx.x;
  const int wave = tid >> 5, lane = tid & 31;
  const int wm = wave & 1, wn = wave >> 1;    // 2 (M) x 4 (N) waves, 64x32 each
  const bool issuer = (__builtin_amdgcn_readfirstlane(tid) < 32);

  v8f acc[4][2];
#pragma unroll
  for (int i = 0; i < 4; ++i)
#pragma unroll
    for (int j = 0; j < 2; ++j) acc[i][j] = (v8f)(0.0f);

  const int row = tid >> 1;            // 128 rows, 2 threads/row
  const int cs  = (tid & 1) * 16;      // 16 K-elements each

  for (int kk = 0; kk < K; kk += 32) {
    __syncthreads();
    // stage A
    if (AF32) {
#pragma unroll
      for (int i = 0; i < 4; ++i) {
        float4 f = *(const float4*)&Af[(size_t)(mb + row) * K + kk + cs + i * 4];
        v4bf h; h[0] = (__bf16)f.x; h[1] = (__bf16)f.y; h[2] = (__bf16)f.z; h[3] = (__bf16)f.w;
        *(v4bf*)&As[row * LDT + cs + i * 4] = h;
      }
    } else {
#ifdef HAVE_TDM
      // TDM: 128x32 bf16 tile, row = 16 DWORDs, pad 4 DWORDs/row -> stride 40
      if (issuer)
        tdm_load_2d((unsigned int)(size_t)&As[0], Ab + (size_t)mb * K + kk,
                    32u, 128u, (unsigned int)K, 3u /*16 DW*/, 3u /*4 DW*/);
#else
#pragma unroll
      for (int i = 0; i < 2; ++i)
        *(v8bf*)&As[row * LDT + cs + i * 8] =
            *(const v8bf*)&Ab[(size_t)(mb + row) * K + kk + cs + i * 8];
#endif
    }
    // stage W (fp32 -> bf16 on VALU)
#pragma unroll
    for (int i = 0; i < 4; ++i) {
      float4 f = *(const float4*)&Wf[(size_t)(nb + row) * K + kk + cs + i * 4];
      v4bf h; h[0] = (__bf16)f.x; h[1] = (__bf16)f.y; h[2] = (__bf16)f.z; h[3] = (__bf16)f.w;
      *(v4bf*)&Bs[row * LDT + cs + i * 4] = h;
    }
    if (kk + 32 < K) {
      __builtin_prefetch(&Wf[(size_t)(nb + row) * K + kk + 32 + cs], 0, 0);
      if (AF32) __builtin_prefetch(&Af[(size_t)(mb + row) * K + kk + 32 + cs], 0, 0);
    }
    if (!AF32 && issuer) tdm_wait();
    __syncthreads();

    v16bf bf[2], af[4];
#pragma unroll
    for (int j = 0; j < 2; ++j) bf[j] = frag_b(&Bs[(wn * 32 + j * 16) * LDT], LDT, lane);
#pragma unroll
    for (int i = 0; i < 4; ++i) af[i] = frag_a(&As[(wm * 64 + i * 16) * LDT], LDT, lane);
#pragma unroll
    for (int i = 0; i < 4; ++i)
#pragma unroll
      for (int j = 0; j < 2; ++j) acc[i][j] = wmma_bf16(af[i], bf[j], acc[i][j]);
  }

  const int r = lane & 15, hf = (lane >> 4) * 8;
#pragma unroll
  for (int i = 0; i < 4; ++i) {
#pragma unroll
    for (int j = 0; j < 2; ++j) {
      int col = nb + wn * 32 + j * 16 + r;
      int row0 = mb + wm * 64 + i * 16 + hf;
      float bv = bias[col];
#pragma unroll
      for (int e = 0; e < 8; ++e) {
        float v = acc[i][j][e] + bv;
        if (OUTF32) Cf[(size_t)(row0 + e) * N + col] = v;
        else        Cb[(size_t)(row0 + e) * N + col] = (__bf16)v;
      }
    }
  }
}

// ---------------- RoPE + pack q/k (D pad 96) and v transposed ----------------
__global__ __launch_bounds__(256) void rope_pack(
    const unsigned short* __restrict__ qkvu, const float* __restrict__ cosp,
    const float* __restrict__ sinp, unsigned short* __restrict__ qbu,
    unsigned short* __restrict__ kbu, unsigned short* __restrict__ vtu) {
  const __bf16* qkv = (const __bf16*)qkvu;
  __bf16* qb = (__bf16*)qbu;
  __bf16* kb = (__bf16*)kbu;
  __bf16* vt = (__bf16*)vtu;
  int idx = blockIdx.x * blockDim.x + threadIdx.x;  // S*H*96
  int d = idx % DPAD;
  int h = (idx / DPAD) % HQ;
  int s = idx / (DPAD * HQ);
  if (s >= S_TOT) return;
  int c = s >> 10, l = s & (LCHUNK - 1);
  float qv = 0.f, kv = 0.f;
  if (d < DH) {
    size_t base = (size_t)s * N_QKV + h * DH;
    float q0 = (float)qkv[base + d];
    float k0 = (float)qkv[base + E_DIM + d];
    int dp = (d < DH / 2) ? d + DH / 2 : d - DH / 2;
    float sgn = (d < DH / 2) ? -1.f : 1.f;
    float qr = (float)qkv[base + dp];
    float kr = (float)qkv[base + E_DIM + dp];
    float cc = cosp[s * DH + d], ss = sinp[s * DH + d];
    qv = q0 * cc + sgn * qr * ss;
    kv = k0 * cc + sgn * kr * ss;
  }
  size_t qkoff = ((size_t)(c * HQ + h) * LCHUNK + l) * DPAD + d;
  qb[qkoff] = (__bf16)(qv * ATT_SCALE);
  kb[qkoff] = (__bf16)kv;
  if (d < DV) {
    float vv = (d < DH) ? (float)qkv[(size_t)s * N_QKV + 2 * E_DIM + h * DH + d] : 0.f;
    vt[((size_t)(c * HQ + h) * DV + d) * LCHUNK + l] = (__bf16)vv;
  }
}

// ---------------- Flash attention: per (chunk*head, 128 q rows) --------------
// Q/K tiles: 96-elem rows, TDM pads 4 DWORDs every 16 DWORDs -> segment layout:
//   elem(row,d) at row*120 + (d>>5)*40 + (d&31)
#define QKL 120
#define SEG 40
#define VLD 136
#define PLD 136
__global__ __launch_bounds__(256) void attn_fa(
    const unsigned short* __restrict__ qbu, const unsigned short* __restrict__ kbu,
    const unsigned short* __restrict__ vtu, unsigned short* __restrict__ attnu) {
  __shared__ __bf16 Qs[128 * QKL];
  __shared__ __bf16 Ks[128 * QKL];
  __shared__ __bf16 Vs[DV * VLD];
  __shared__ __bf16 Ps[128 * PLD];
  const __bf16* qb = (const __bf16*)qbu;
  const __bf16* kb = (const __bf16*)kbu;
  const __bf16* vt = (const __bf16*)vtu;
  __bf16* attn = (__bf16*)attnu;

  const int ch = blockIdx.x;           // chunk*16 + head
  const int q0 = blockIdx.y * 128;
  const int tid = threadIdx.x, wave = tid >> 5, lane = tid & 31;
  const int r = lane & 15, hf = (lane >> 4) * 8;
  const bool issuer = (__builtin_amdgcn_readfirstlane(tid) < 32);
  const __bf16* Qg = qb + (size_t)ch * LCHUNK * DPAD;
  const __bf16* Kg = kb + (size_t)ch * LCHUNK * DPAD;
  const __bf16* Vg = vt + (size_t)ch * DV * LCHUNK;

#ifdef HAVE_TDM
  if (issuer)
    tdm_load_2d((unsigned int)(size_t)&Qs[0], Qg + (size_t)q0 * DPAD,
                DPAD, 128u, DPAD, 3u, 3u);
#else
  for (int i = tid; i < 128 * 12; i += 256) {
    int row = i / 12, c8 = i % 12;
    *(v8bf*)&Qs[row * QKL + (c8 >> 2) * SEG + (c8 & 3) * 8] =
        *(const v8bf*)&Qg[(size_t)(q0 + row) * DPAD + c8 * 8];
  }
#endif

  float m_i[8], l_i[8];
  v8f oacc[5];
#pragma unroll
  for (int e = 0; e < 8; ++e) { m_i[e] = -1e30f; l_i[e] = 0.f; }
#pragma unroll
  for (int t = 0; t < 5; ++t) oacc[t] = (v8f)(0.0f);

  for (int kv0 = 0; kv0 < LCHUNK; kv0 += 128) {
    __syncthreads();
#ifdef HAVE_TDM
    if (issuer) {
      tdm_load_2d((unsigned int)(size_t)&Ks[0], Kg + (size_t)kv0 * DPAD,
                  DPAD, 128u, DPAD, 3u, 3u);
      // V tile: 128-elem rows = 64 DWORDs, pad 4 DWORDs at row end -> stride 136
      tdm_load_2d((unsigned int)(size_t)&Vs[0], Vg + kv0,
                  128u, DV, LCHUNK, 5u /*64 DW*/, 3u /*4 DW*/);
      tdm_wait();
    }
#else
    for (int i = tid; i < 128 * 12; i += 256) {
      int row = i / 12, c8 = i % 12;
      *(v8bf*)&Ks[row * QKL + (c8 >> 2) * SEG + (c8 & 3) * 8] =
          *(const v8bf*)&Kg[(size_t)(kv0 + row) * DPAD + c8 * 8];
    }
    for (int i = tid; i < DV * 16; i += 256) {
      int row = i / 16, seg = i % 16;
      *(v8bf*)&Vs[row * VLD + seg * 8] = *(const v8bf*)&Vg[(size_t)row * LCHUNK + kv0 + seg * 8];
    }
#endif
    __syncthreads();

    // S = Q K^T : wave owns 16 rows x 128 cols (8 tiles), K-dim = 96
    v8f sc[8];
#pragma unroll
    for (int t = 0; t < 8; ++t) sc[t] = (v8f)(0.0f);
#pragma unroll
    for (int ks = 0; ks < DPAD; ks += 32) {
      const int so = (ks >> 5) * SEG;
      v16bf aq = frag_a(&Qs[(wave * 16) * QKL + so], QKL, lane);
#pragma unroll
      for (int g = 0; g < 2; ++g) {
        v16bf bk[4];
#pragma unroll
        for (int t = 0; t < 4; ++t)
          bk[t] = frag_b(&Ks[((g * 4 + t) * 16) * QKL + so], QKL, lane);
#pragma unroll
        for (int t = 0; t < 4; ++t) sc[g * 4 + t] = wmma_bf16(aq, bk[t], sc[g * 4 + t]);
      }
    }

    // online softmax: lane elem e -> row = wave*16 + hf + e, col = t*16 + r
    float pm[8];
#pragma unroll
    for (int e = 0; e < 8; ++e) {
      float mx = sc[0][e];
#pragma unroll
      for (int t = 1; t < 8; ++t) mx = fmaxf(mx, sc[t][e]);
      pm[e] = mx;
    }
#pragma unroll
    for (int e = 0; e < 8; ++e)
#pragma unroll
      for (int msk = 1; msk <= 8; msk <<= 1)
        pm[e] = fmaxf(pm[e], __shfl_xor(pm[e], msk, 32));

    float fac[8], psum[8];
#pragma unroll
    for (int e = 0; e < 8; ++e) {
      float mn = fmaxf(m_i[e], pm[e]);
      fac[e] = __expf(m_i[e] - mn);
      m_i[e] = mn;
      psum[e] = 0.f;
    }
#pragma unroll
    for (int t = 0; t < 8; ++t)
#pragma unroll
      for (int e = 0; e < 8; ++e) {
        float p = __expf(sc[t][e] - m_i[e]);
        psum[e] += p;
        Ps[(wave * 16 + hf + e) * PLD + t * 16 + r] = (__bf16)p;
      }
#pragma unroll
    for (int e = 0; e < 8; ++e)
#pragma unroll
      for (int msk = 1; msk <= 8; msk <<= 1)
        psum[e] += __shfl_xor(psum[e], msk, 32);
#pragma unroll
    for (int e = 0; e < 8; ++e) l_i[e] = l_i[e] * fac[e] + psum[e];
#pragma unroll
    for (int t = 0; t < 5; ++t)
#pragma unroll
      for (int e = 0; e < 8; ++e) oacc[t][e] *= fac[e];
    __syncthreads();

    // O += P V : K-dim = 128, N = 80 (5 tiles)
#pragma unroll
    for (int ks = 0; ks < 128; ks += 32) {
      v16bf ap = frag_a(&Ps[(wave * 16) * PLD + ks], PLD, lane);
      v16bf bv[5];
#pragma unroll
      for (int t = 0; t < 5; ++t) bv[t] = frag_b(&Vs[(t * 16) * VLD + ks], VLD, lane);
#pragma unroll
      for (int t = 0; t < 5; ++t) oacc[t] = wmma_bf16(ap, bv[t], oacc[t]);
    }
  }

  const int c = ch >> 4, h = ch & 15;
#pragma unroll
  for (int t = 0; t < 5; ++t) {
    int d = t * 16 + r;
    if (d < DH) {
#pragma unroll
      for (int e = 0; e < 8; ++e) {
        int s = c * LCHUNK + q0 + wave * 16 + hf + e;
        attn[(size_t)s * E_DIM + h * DH + d] = (__bf16)(oacc[t][e] / l_i[e]);
      }
    }
  }
}

// -----------------------------------------------------------------------------
extern "C" void kernel_launch(void* const* d_in, const int* in_sizes, int n_in,
                              void* d_out, int out_size, void* d_ws, size_t ws_size,
                              hipStream_t stream) {
  const float* hidden = (const float*)d_in[0];
  const float* cosp   = (const float*)d_in[1];
  const float* sinp   = (const float*)d_in[2];
  const float* qkv_w  = (const float*)d_in[3];
  const float* qkv_b  = (const float*)d_in[4];
  const float* proj_w = (const float*)d_in[5];
  const float* proj_b = (const float*)d_in[6];
  float* out = (float*)d_out;

  char* ws = (char*)d_ws;
  size_t off = 0;
  unsigned short* qkv_bf = (unsigned short*)(ws + off); off += (size_t)S_TOT * N_QKV * 2;
  unsigned short* qb     = (unsigned short*)(ws + off); off += (size_t)4 * HQ * LCHUNK * DPAD * 2;
  unsigned short* kbp    = (unsigned short*)(ws + off); off += (size_t)4 * HQ * LCHUNK * DPAD * 2;
  unsigned short* vt     = (unsigned short*)(ws + off); off += (size_t)4 * HQ * DV * LCHUNK * 2;
  unsigned short* attnb  = (unsigned short*)(ws + off); off += (size_t)S_TOT * E_DIM * 2;

  // 1) QKV = hidden @ qkv_w^T + qkv_b  (bf16 out)
  gemm_wmma<true, false><<<dim3(S_TOT / 128, N_QKV / 128), 256, 0, stream>>>(
      hidden, nullptr, qkv_w, qkv_b, nullptr, qkv_bf, S_TOT, N_QKV, E_DIM);

  // 2) RoPE + pack (q scaled by 72^-0.5, v transposed)
  int rope_threads = S_TOT * HQ * DPAD;
  rope_pack<<<(rope_threads + 255) / 256, 256, 0, stream>>>(qkv_bf, cosp, sinp, qb, kbp, vt);

  // 3) flash attention per (chunk*head, 128-row q block)
  attn_fa<<<dim3(4 * HQ, LCHUNK / 128), 256, 0, stream>>>(qb, kbp, vt, attnb);

  // 4) out = attn @ proj_w^T + proj_b  (fp32 out)
  gemm_wmma<false, true><<<dim3(S_TOT / 128, E_DIM / 128), 256, 0, stream>>>(
      nullptr, attnb, proj_w, proj_b, out, nullptr, S_TOT, E_DIM, E_DIM);
}
